// LocalAggregator_53618371723802
// MI455X (gfx1250) — matrix operand order repaired and checked by reading, not compiled, not run on hardware
//
#include <hip/hip_runtime.h>

typedef __attribute__((ext_vector_type(16))) _Float16 v16h;
typedef __attribute__((ext_vector_type(8)))  float    v8f;

#define GRID_SZ   0.4f
#define INV_GRID  2.5f
#define MULTS     3.0f
#define PCMIN_X  (-40.0f)
#define PCMIN_Y  (-40.0f)
#define PCMIN_Z  (-1.0f)
#define SEM_PAD   264          // LDS row stride (halves) -> distinct banks per class
#define NEG_HALF_LOG2E (-0.72134752044448170368f)   // -0.5 * log2(e)

// ---------------------------------------------------------------------------
// Per-gaussian precompute, packed 16 floats (64B):
//   [0..5]  conic pre-scaled by -0.5*log2(e), cross terms doubled:
//           ah00, ah01*2, ah02*2, ah11, ah12*2, ah22
//   [6..8]  mean xyz     [9] opacity
//   [10..12] voxel box lo (int bits)   [13] 2*radius (int bits)
// Inner loop then computes the v_exp_f32 argument directly and masks with an
// unsigned range check: (unsigned)(p - lo) <= 2r per axis.
// ---------------------------------------------------------------------------
__global__ void la_prep_gauss(const float* __restrict__ means,
                              const float* __restrict__ opac,
                              const float* __restrict__ scales,
                              const float* __restrict__ cov,
                              float* __restrict__ gp, int M) {
  int g = blockIdx.x * blockDim.x + threadIdx.x;
  if (g >= M) return;
  const float* c = cov + (size_t)g * 9;
  float c00 = c[0], c01 = c[1], c02 = c[2];
  float c11 = c[4], c12 = c[5], c22 = c[8];
  float M00 = c11 * c22 - c12 * c12;
  float M01 = c02 * c12 - c01 * c22;
  float M02 = c01 * c12 - c02 * c11;
  float det = fmaf(c00, M00, fmaf(c01, M01, c02 * M02));
  float inv = 1.0f / det;
  float A00 = M00 * inv, A01 = M01 * inv, A02 = M02 * inv;
  float A11 = (c00 * c22 - c02 * c02) * inv;
  float A12 = (c01 * c02 - c00 * c12) * inv;
  float A22 = (c00 * c11 - c01 * c01) * inv;
  float mx = means[g * 3 + 0], my = means[g * 3 + 1], mz = means[g * 3 + 2];
  int mix = (int)floorf((mx - PCMIN_X) * INV_GRID);
  int miy = (int)floorf((my - PCMIN_Y) * INV_GRID);
  int miz = (int)floorf((mz - PCMIN_Z) * INV_GRID);
  float s0 = scales[g * 3 + 0], s1 = scales[g * 3 + 1], s2 = scales[g * 3 + 2];
  float smax = fmaxf(s0, fmaxf(s1, s2));
  int rad = (int)ceilf(smax * MULTS * INV_GRID);
  const float k = NEG_HALF_LOG2E;
  float* o = gp + (size_t)g * 16;
  o[0] = k * A00;        o[1] = 2.0f * k * A01; o[2] = 2.0f * k * A02;
  o[3] = k * A11;        o[4] = 2.0f * k * A12; o[5] = k * A22;
  o[6] = mx;  o[7] = my;  o[8] = mz;  o[9] = opac[g];
  o[10] = __int_as_float(mix - rad);
  o[11] = __int_as_float(miy - rad);
  o[12] = __int_as_float(miz - rad);
  o[13] = __int_as_float(2 * rad);
  o[14] = 0.0f; o[15] = 0.0f;
}

// ---------------------------------------------------------------------------
// semantics [M,C] f32 -> semT [32][M] f16 (classes >= C zero padded).
// ---------------------------------------------------------------------------
__global__ void la_prep_sem(const float* __restrict__ sem,
                            _Float16* __restrict__ semT, int M, int C) {
  int idx = blockIdx.x * blockDim.x + threadIdx.x;
  if (idx >= 32 * M) return;
  int cc = idx / M;
  int m  = idx - cc * M;
  float v = (cc < C) ? sem[(size_t)m * C + cc] : 0.0f;
  semT[(size_t)cc * M + m] = (_Float16)v;
}

// ---------------------------------------------------------------------------
// Main aggregation. Block = 256 threads = 8 waves; wave owns 16 points.
// Per 256-gaussian chunk: stage gaussian structs (16KB) + semantics slab
// (16.5KB) in LDS; per 32-gaussian K-step: ds-load both B fragments first,
// build the w tile (branchless exp2 + unsigned box mask), then two
// v_wmma_f32_16x16x32_f16 into f32 accumulators.
// ---------------------------------------------------------------------------
__global__ __launch_bounds__(256)
void la_main(const float* __restrict__ pts,
             const float* __restrict__ gp,
             const _Float16* __restrict__ semT,
             float* __restrict__ out,
             int N, int M, int C) {
  __shared__ float    sg[256 * 16];      // 16 KB gaussian params
  __shared__ _Float16 ss[32 * SEM_PAD];  // 16.5 KB semantics chunk

  const int tid   = threadIdx.x;
  const int wave  = tid >> 5;
  const int lane  = tid & 31;
  const int row   = lane & 15;     // A row / B column (class)
  const int khalf = lane >> 4;     // K half held by this lane
  const int pbase = blockIdx.x * 128 + wave * 16;
  const int pt    = pbase + row;

  const float px = pts[pt * 3 + 0];
  const float py = pts[pt * 3 + 1];
  const float pz = pts[pt * 3 + 2];
  const int pix = (int)floorf((px - PCMIN_X) * INV_GRID);
  const int piy = (int)floorf((py - PCMIN_Y) * INV_GRID);
  const int piz = (int)floorf((pz - PCMIN_Z) * INV_GRID);

  v8f acc0 = {};   // classes 0..15
  v8f acc1 = {};   // classes 16..31 (16,17 real)
  union { v16h v; _Float16 e[16]; } af;
  union { v16h v; float4 f[2]; } bf0, bf1;

  const int srow  = tid >> 3;  // 0..31  (class row staged by this thread)
  const int spart = tid & 7;   // 0..7   (32-half segment)

  for (int chunk = 0; chunk < M; chunk += 256) {
    __syncthreads();
    {
      const float4* gsrc = (const float4*)(gp + (size_t)(chunk + tid) * 16);
      float4* gdst = (float4*)(&sg[tid * 16]);
      gdst[0] = gsrc[0]; gdst[1] = gsrc[1]; gdst[2] = gsrc[2]; gdst[3] = gsrc[3];

      const float4* ssrc = (const float4*)(semT + (size_t)srow * M + chunk + spart * 32);
      float4* sdst = (float4*)(ss + srow * SEM_PAD + spart * 32);
      sdst[0] = ssrc[0]; sdst[1] = ssrc[1]; sdst[2] = ssrc[2]; sdst[3] = ssrc[3];

      if (chunk + 256 < M) {
        __builtin_prefetch(gp + (size_t)(chunk + 256 + tid) * 16, 0, 3);
        __builtin_prefetch(semT + (size_t)srow * M + chunk + 256 + spart * 32, 0, 3);
      }
    }
    __syncthreads();

    for (int kb = 0; kb < 256; kb += 32) {
      // ---- B fragments first (latency hidden by the 16-element VALU chain)
      const _Float16* bp  = ss + row * SEM_PAD + kb + khalf * 16;
      const _Float16* bp1 = bp + 16 * SEM_PAD;
      bf0.f[0] = *(const float4*)(bp);
      bf0.f[1] = *(const float4*)(bp + 8);
      bf1.f[0] = *(const float4*)(bp1);
      bf1.f[1] = *(const float4*)(bp1 + 8);

      // ---- A fragment: w for this lane's 16 (point,gaussian) pairs
#pragma unroll
      for (int e = 0; e < 16; ++e) {
        const int K = e + 16 * khalf;               // contiguous k = 16*khalf + e
        const float* s = &sg[(kb + K) * 16];
        const float4 u0 = *(const float4*)(s);       // ah00 ah01x2 ah02x2 ah11
        const float4 u1 = *(const float4*)(s + 4);   // ah12x2 ah22 mx my
        const float4 u2 = *(const float4*)(s + 8);   // mz op lox loy
        const float2 u3 = *(const float2*)(s + 12);  // loz tworad
        float d0 = px - u1.z;
        float d1 = py - u1.w;
        float d2 = pz - u2.x;
        float q = d0 * (u0.x * d0 + u0.y * d1 + u0.z * d2)
                + d1 * (u0.w * d1 + u1.x * d2)
                + u1.y * d2 * d2;                    // = -0.5*log2e * dAd
        unsigned v0 = (unsigned)(pix - __float_as_int(u2.z));
        unsigned v1 = (unsigned)(piy - __float_as_int(u2.w));
        unsigned v2 = (unsigned)(piz - __float_as_int(u3.x));
        unsigned um = v0 > v1 ? v0 : v1;
        um = um > v2 ? um : v2;
        q = (um <= (unsigned)__float_as_int(u3.y)) ? q : -3.0e38f;  // cndmask
        af.e[e] = (_Float16)(u2.y * __builtin_amdgcn_exp2f(q));
      }

      acc0 = __builtin_amdgcn_wmma_f32_16x16x32_f16(
          false, af.v, false, bf0.v, (short)0, acc0, false, false);
      acc1 = __builtin_amdgcn_wmma_f32_16x16x32_f16(
          false, af.v, false, bf1.v, (short)0, acc1, false, false);
    }
  }

  // ---- D layout: VGPR j -> row j + 8*khalf, column = lane&15
  const int cls = row;
#pragma unroll
  for (int j = 0; j < 8; ++j) {
    int prow = j + khalf * 8;
    out[(size_t)(pbase + prow) * C + cls] = acc0[j];
  }
  if (cls < C - 16) {
#pragma unroll
    for (int j = 0; j < 8; ++j) {
      int prow = j + khalf * 8;
      out[(size_t)(pbase + prow) * C + 16 + cls] = acc1[j];
    }
  }
}

// ---------------------------------------------------------------------------
extern "C" void kernel_launch(void* const* d_in, const int* in_sizes, int n_in,
                              void* d_out, int out_size, void* d_ws, size_t ws_size,
                              hipStream_t stream) {
  const float* pts    = (const float*)d_in[0];   // [1,N,3]
  const float* means  = (const float*)d_in[1];   // [1,M,3]
  const float* opac   = (const float*)d_in[2];   // [1,M]
  const float* sem    = (const float*)d_in[3];   // [1,M,C]
  const float* scales = (const float*)d_in[4];   // [1,M,3]
  const float* cov    = (const float*)d_in[5];   // [1,M,3,3]
  float* out = (float*)d_out;

  const int N = in_sizes[0] / 3;
  const int M = in_sizes[2];
  const int C = in_sizes[3] / M;

  float*    gp   = (float*)d_ws;                                              // M*16 f32
  _Float16* semT = (_Float16*)((char*)d_ws + (size_t)M * 16 * sizeof(float)); // 32*M f16

  la_prep_gauss<<<(M + 255) / 256, 256, 0, stream>>>(means, opac, scales, cov, gp, M);
  la_prep_sem<<<(32 * M + 255) / 256, 256, 0, stream>>>(sem, semT, M, C);
  la_main<<<N / 128, 256, 0, stream>>>(pts, gp, semT, out, N, M, C);
}
